// MambaFormerCrossAttentionLayer_66288525247129
// MI455X (gfx1250) — matrix operand order repaired and checked by reading, not compiled, not run on hardware
//
#include <hip/hip_runtime.h>
#include <hip/hip_bf16.h>
#include <math.h>

typedef __attribute__((ext_vector_type(16))) _Float16 v16h;
typedef __attribute__((ext_vector_type(8)))  _Float16 v8h;
typedef __attribute__((ext_vector_type(8)))  float    v8f;

#define L_SEQ   4096
#define DMODEL  1024
#define DINNER  2048
#define DSTATE  16
#define DTRANK  64

// =====================================================================
// Big-tile GEMM: C[M,N] = A[M,K](f16) x Bt[K,N](f16, K-major) + bias[N]
// wave tile 32Mx64N (2 A-tiles x 4 B-tiles = 8 WMMAs / K-step),
// block = 8 waves as 2Mx4N groups -> 64M x 256N block tile.
// Requires M%64==0, N%256==0, K%32==0.
// Loads per K-step: 4 (A) b128 + 64 (B) 16-bit scalar for 8 WMMAs -> 8.5 loads/WMMA.
// =====================================================================
__global__ __launch_bounds__(256) void k_gemm_f16_big(
    const _Float16* __restrict__ A, const _Float16* __restrict__ Bt,
    const float* __restrict__ bias, float* __restrict__ C,
    int M, int N, int K)
{
  const int wave = threadIdx.x >> 5;
  const int lane = threadIdx.x & 31;
  const int wm = wave >> 2;             // 0..1
  const int wn = wave & 3;              // 0..3
  const int m0 = blockIdx.y * 64 + wm * 32;
  const int n0 = blockIdx.x * 256 + wn * 64;

  const int khalf = lane >> 4;
  const _Float16* a_base0 = A + (size_t)(m0 + (lane & 15)) * K + khalf * 8;
  const _Float16* a_base1 = a_base0 + (size_t)16 * K;
  const _Float16* b_base  = Bt + (size_t)(khalf * 8) * N + n0 + (lane & 15);

  v8f acc[2][4] = {};
  for (int k0 = 0; k0 < K; k0 += 32) {
    v8h a0lo = *(const v8h*)(a_base0 + k0);
    v8h a0hi = *(const v8h*)(a_base0 + k0 + 16);
    v8h a1lo = *(const v8h*)(a_base1 + k0);
    v8h a1hi = *(const v8h*)(a_base1 + k0 + 16);
    v16h a0, a1;
#pragma unroll
    for (int i = 0; i < 8; ++i) {
      a0[i] = a0lo[i]; a0[8 + i] = a0hi[i];
      a1[i] = a1lo[i]; a1[8 + i] = a1hi[i];
    }
    const _Float16* brow = b_base + (size_t)k0 * N;
#pragma unroll
    for (int j = 0; j < 4; ++j) {
      v16h b;
#pragma unroll
      for (int s = 0; s < 8; ++s) {
        b[s] = brow[(size_t)s * N + 16 * j];
        b[8 + s] = brow[(size_t)(16 + s) * N + 16 * j];
      }
      acc[0][j] = __builtin_amdgcn_wmma_f32_16x16x32_f16(false, a0, false, b, (short)0, acc[0][j], false, false);
      acc[1][j] = __builtin_amdgcn_wmma_f32_16x16x32_f16(false, a1, false, b, (short)0, acc[1][j], false, false);
    }
  }
  const int nn = lane & 15;
  const int mh = lane >> 4;
#pragma unroll
  for (int j = 0; j < 4; ++j) {
    const float bv = bias ? bias[n0 + 16 * j + nn] : 0.f;
#pragma unroll
    for (int i = 0; i < 8; ++i) {
      C[(size_t)(m0 + 8 * mh + i) * N + n0 + 16 * j + nn]      = acc[0][j][i] + bv;
      C[(size_t)(m0 + 16 + 8 * mh + i) * N + n0 + 16 * j + nn] = acc[1][j][i] + bv;
    }
  }
}

// =====================================================================
// Small-N GEMM (only x_proj, N=96): wave tile 16Mx32N, 8 waves in M.
// Requires M%128==0, N%32==0, K%32==0.
// =====================================================================
__global__ __launch_bounds__(256) void k_gemm_f16(
    const _Float16* __restrict__ A, const _Float16* __restrict__ Bt,
    const float* __restrict__ bias, float* __restrict__ C,
    int M, int N, int K)
{
  const int wave = threadIdx.x >> 5;
  const int lane = threadIdx.x & 31;
  const int m0 = (blockIdx.y * 8 + wave) * 16;
  const int n0 = blockIdx.x * 32;

  const int khalf = lane >> 4;
  const _Float16* a_base = A + (size_t)(m0 + (lane & 15)) * K + khalf * 8;
  const _Float16* b_base = Bt + (size_t)(khalf * 8) * N + n0 + (lane & 15);

  v8f acc0 = {}; v8f acc1 = {};
  for (int k0 = 0; k0 < K; k0 += 32) {
    v8h alo = *(const v8h*)(a_base + k0);
    v8h ahi = *(const v8h*)(a_base + k0 + 16);
    v16h a;
#pragma unroll
    for (int i = 0; i < 8; ++i) { a[i] = alo[i]; a[8 + i] = ahi[i]; }
    const _Float16* brow = b_base + (size_t)k0 * N;
    v16h b0, b1;
#pragma unroll
    for (int s = 0; s < 8; ++s) {
      b0[s] = brow[(size_t)s * N];      b0[8 + s] = brow[(size_t)(16 + s) * N];
      b1[s] = brow[(size_t)s * N + 16]; b1[8 + s] = brow[(size_t)(16 + s) * N + 16];
    }
    acc0 = __builtin_amdgcn_wmma_f32_16x16x32_f16(false, a, false, b0, (short)0, acc0, false, false);
    acc1 = __builtin_amdgcn_wmma_f32_16x16x32_f16(false, a, false, b1, (short)0, acc1, false, false);
  }
  const int nn = lane & 15;
  const int mh = lane >> 4;
  const float bv0 = bias ? bias[n0 + nn]      : 0.f;
  const float bv1 = bias ? bias[n0 + 16 + nn] : 0.f;
#pragma unroll
  for (int i = 0; i < 8; ++i) {
    const int m = m0 + 8 * mh + i;
    C[(size_t)m * N + n0 + nn]      = acc0[i] + bv0;
    C[(size_t)m * N + n0 + 16 + nn] = acc1[i] + bv1;
  }
}

// W (N,K) f32 -> Wt (K,N) f16 (K-major for WMMA B operand)
__global__ void k_transpose_f16(const float* __restrict__ W,
                                _Float16* __restrict__ Wt, int N, int K)
{
  size_t i = (size_t)blockIdx.x * blockDim.x + threadIdx.x;
  if (i >= (size_t)N * K) return;
  int k = (int)(i / N), n = (int)(i % N);
  Wt[i] = (_Float16)W[(size_t)n * K + k];
}

// strided f32 -> f16 (row stride S, take C cols from col 0)
__global__ void k_cvt_f16(const float* __restrict__ in, _Float16* __restrict__ out,
                          int M, int S, int C)
{
  size_t i = (size_t)blockIdx.x * blockDim.x + threadIdx.x;
  if (i >= (size_t)M * C) return;
  int r = (int)(i / C), c = (int)(i % C);
  out[i] = (_Float16)in[(size_t)r * S + c];
}

// depthwise causal conv (D_CONV=4) + SiLU; reads xc = xz[:, 0:2048]
__global__ void k_conv_silu(const float* __restrict__ xz,
                            const float* __restrict__ cw, const float* __restrict__ cb,
                            float* __restrict__ xs, _Float16* __restrict__ xsh)
{
  size_t i = (size_t)blockIdx.x * blockDim.x + threadIdx.x;
  if (i >= (size_t)L_SEQ * DINNER) return;
  int t = (int)(i / DINNER), d = (int)(i % DINNER);
  float acc = cb[d];
#pragma unroll
  for (int k = 0; k < 4; ++k) {
    int tt = t - 3 + k;
    if (tt >= 0) acc += cw[d * 4 + k] * xz[(size_t)tt * (2 * DINNER) + d];
  }
  float s = acc / (1.f + __expf(-acc));
  xs[i] = s; xsh[i] = (_Float16)s;
}

__global__ void k_softplus(float* __restrict__ x, size_t n)
{
  size_t i = (size_t)blockIdx.x * blockDim.x + threadIdx.x;
  if (i >= n) return;
  float v = x[i];
  x[i] = (v > 20.f) ? v : log1pf(__expf(v));
}

// selective scan: one thread per (d,n); 16-lane shuffle reduction over states
__global__ __launch_bounds__(256) void k_scan(
    const float* __restrict__ xs, const float* __restrict__ dt,
    const float* __restrict__ xdbl, const float* __restrict__ A_log,
    const float* __restrict__ Dv, float* __restrict__ ys)
{
  int tid = blockIdx.x * 256 + threadIdx.x;   // 0 .. 32767
  int d = tid >> 4, n = tid & 15;
  float Adn = -__expf(A_log[d * DSTATE + n]);
  float Dd  = Dv[d];
  float h = 0.f;
  for (int t = 0; t < L_SEQ; ++t) {
    float dtv = dt[(size_t)t * DINNER + d];
    float xv  = xs[(size_t)t * DINNER + d];
    float Bv  = xdbl[(size_t)t * 96 + DTRANK + n];
    float Cv  = xdbl[(size_t)t * 96 + DTRANK + DSTATE + n];
    h = __expf(dtv * Adn) * h + (dtv * xv) * Bv;
    float p = h * Cv;
    p += __shfl_xor(p, 1, 32);
    p += __shfl_xor(p, 2, 32);
    p += __shfl_xor(p, 4, 32);
    p += __shfl_xor(p, 8, 32);
    if (n == 0) ys[(size_t)t * DINNER + d] = p + Dd * xv;
  }
}

// gated = ys * silu(z);  z = xz[:, 2048:4096]
__global__ void k_gate_silu(const float* __restrict__ ys, const float* __restrict__ xz,
                            _Float16* __restrict__ g)
{
  size_t i = (size_t)blockIdx.x * blockDim.x + threadIdx.x;
  if (i >= (size_t)L_SEQ * DINNER) return;
  int t = (int)(i / DINNER), d = (int)(i % DINNER);
  float z = xz[(size_t)t * (2 * DINNER) + DINNER + d];
  float s = z / (1.f + __expf(-z));
  g[i] = (_Float16)(ys[i] * s);
}

__global__ void k_gelu(const float* __restrict__ in, _Float16* __restrict__ outh, size_t n)
{
  size_t i = (size_t)blockIdx.x * blockDim.x + threadIdx.x;
  if (i >= n) return;
  float x = in[i];
  outh[i] = (_Float16)(0.5f * x * (1.f + erff(x * 0.70710678118654752f)));
}

// out = LN(a + b), a broadcast over rows if abroad; optional f16 copy
__global__ __launch_bounds__(256) void k_add_ln(
    const float* __restrict__ a, const float* __restrict__ b,
    const float* __restrict__ g, const float* __restrict__ be,
    float* __restrict__ out, _Float16* __restrict__ outh, int D, int abroad)
{
  int row = blockIdx.x;
  float s = 0.f, s2 = 0.f;
  for (int j = threadIdx.x; j < D; j += 256) {
    float v = (abroad ? a[j] : a[(size_t)row * D + j]) + b[(size_t)row * D + j];
    s += v; s2 += v * v;
  }
#pragma unroll
  for (int o = 16; o >= 1; o >>= 1) { s += __shfl_xor(s, o, 32); s2 += __shfl_xor(s2, o, 32); }
  __shared__ float ps[8], ps2[8];
  if ((threadIdx.x & 31) == 0) { ps[threadIdx.x >> 5] = s; ps2[threadIdx.x >> 5] = s2; }
  __syncthreads();
  float ts = 0.f, ts2 = 0.f;
#pragma unroll
  for (int w = 0; w < 8; ++w) { ts += ps[w]; ts2 += ps2[w]; }
  float mu  = ts / D;
  float var = ts2 / D - mu * mu;
  float inv = rsqrtf(var + 1e-5f);
  for (int j = threadIdx.x; j < D; j += 256) {
    float v = (abroad ? a[j] : a[(size_t)row * D + j]) + b[(size_t)row * D + j];
    float o = (v - mu) * inv * g[j] + be[j];
    out[(size_t)row * D + j] = o;
    if (outh) outh[(size_t)row * D + j] = (_Float16)o;
  }
}

__global__ void k_gemv(const float* __restrict__ W, const float* __restrict__ x,
                       const float* __restrict__ b, float* __restrict__ out, int N, int K)
{
  int n = blockIdx.x * blockDim.x + threadIdx.x;
  if (n >= N) return;
  float acc = b ? b[n] : 0.f;
  for (int k = 0; k < K; ++k) acc += W[(size_t)n * K + k] * x[k];
  out[n] = acc;
}

__global__ void k_store_one(float* p, float v) { *p = v; }

// =====================================================================
extern "C" void kernel_launch(void* const* d_in, const int* in_sizes, int n_in,
                              void* d_out, int out_size, void* d_ws, size_t ws_size,
                              hipStream_t stream)
{
  const float* emb    = (const float*)d_in[0];
  const float* clin   = (const float*)d_in[1];
  const float* w_in   = (const float*)d_in[2];
  const float* conv_w = (const float*)d_in[3];
  const float* conv_b = (const float*)d_in[4];
  const float* w_xp   = (const float*)d_in[5];
  const float* w_dt   = (const float*)d_in[6];
  const float* b_dt   = (const float*)d_in[7];
  const float* A_log  = (const float*)d_in[8];
  const float* Dvec   = (const float*)d_in[9];
  const float* w_out  = (const float*)d_in[10];
  const float* ln1_g  = (const float*)d_in[11];
  const float* ln1_b  = (const float*)d_in[12];
  const float* w_ain  = (const float*)d_in[13];
  const float* b_ain  = (const float*)d_in[14];
  const float* w_aout = (const float*)d_in[15];
  const float* b_aout = (const float*)d_in[16];
  const float* ln2_g  = (const float*)d_in[17];
  const float* ln2_b  = (const float*)d_in[18];
  const float* w_f1   = (const float*)d_in[19];
  const float* b_f1   = (const float*)d_in[20];
  const float* w_f2   = (const float*)d_in[21];
  const float* b_f2   = (const float*)d_in[22];
  const float* ln3_g  = (const float*)d_in[23];
  const float* ln3_b  = (const float*)d_in[24];
  float* out = (float*)d_out;

  // ---- workspace bump allocator (256B aligned) ----
  char* ws = (char*)d_ws;
  size_t off = 0;
  auto alloc = [&](size_t bytes) -> char* {
    char* p = ws + off;
    off = (off + bytes + 255) & ~(size_t)255;
    return p;
  };
  _Float16* wt_in  = (_Float16*)alloc((size_t)DMODEL * (2 * DINNER) * 2); // 1024x4096
  _Float16* wt_xp  = (_Float16*)alloc((size_t)DINNER * 96 * 2);           // 2048x96
  _Float16* wt_dt  = (_Float16*)alloc((size_t)DTRANK * DINNER * 2);       // 64x2048
  _Float16* wt_out = (_Float16*)alloc((size_t)DINNER * DMODEL * 2);       // 2048x1024
  _Float16* wt_f1  = (_Float16*)alloc((size_t)DMODEL * (2 * DMODEL) * 2); // 1024x2048
  _Float16* wt_f2  = (_Float16*)alloc((size_t)(2 * DMODEL) * DMODEL * 2); // 2048x1024
  char* hbufA = alloc((size_t)L_SEQ * DMODEL * 2);      // h_emb, later h_seq2
  float* xz   = (float*)alloc((size_t)L_SEQ * 2 * DINNER * 4);
  char* fbufA = alloc((size_t)L_SEQ * DINNER * 4);      // xs_f32, later ffn_lin1
  char* hbufB = alloc((size_t)L_SEQ * DINNER * 2);      // xs_f16, later h_gated
  float* xdbl = (float*)alloc((size_t)L_SEQ * 96 * 4);
  _Float16* h_dtin = (_Float16*)alloc((size_t)L_SEQ * DTRANK * 2);
  char* fbufB = alloc((size_t)L_SEQ * DINNER * 4);      // dt, later mamba_out|seq1
  char* fbufC = alloc((size_t)L_SEQ * DINNER * 4);      // ys, later h_ffn|ffn_out
  float* seq2 = (float*)alloc((size_t)L_SEQ * DMODEL * 4);
  float* vvec = (float*)alloc(DMODEL * 4);
  float* avec = (float*)alloc(DMODEL * 4);

  const size_t LD  = (size_t)L_SEQ * DINNER;   // 8.4M
  const size_t LM  = (size_t)L_SEQ * DMODEL;   // 4.2M
  const int TB = 256;
  auto blocks = [](size_t n) { return (unsigned)((n + 255) / 256); };

  // ---- weight transposes (f32 -> K-major f16) ----
  k_transpose_f16<<<blocks((size_t)(2*DINNER)*DMODEL), TB, 0, stream>>>(w_in,  wt_in,  2*DINNER, DMODEL);
  k_transpose_f16<<<blocks((size_t)96*DINNER),        TB, 0, stream>>>(w_xp,  wt_xp,  96,       DINNER);
  k_transpose_f16<<<blocks((size_t)DINNER*DTRANK),    TB, 0, stream>>>(w_dt,  wt_dt,  DINNER,   DTRANK);
  k_transpose_f16<<<blocks((size_t)DMODEL*DINNER),    TB, 0, stream>>>(w_out, wt_out, DMODEL,   DINNER);
  k_transpose_f16<<<blocks((size_t)(2*DMODEL)*DMODEL),TB, 0, stream>>>(w_f1,  wt_f1,  2*DMODEL, DMODEL);
  k_transpose_f16<<<blocks((size_t)DMODEL*2*DMODEL),  TB, 0, stream>>>(w_f2,  wt_f2,  DMODEL,   2*DMODEL);

  // ---- Mamba branch ----
  _Float16* h_emb = (_Float16*)hbufA;
  k_cvt_f16<<<blocks(LM), TB, 0, stream>>>(emb, h_emb, L_SEQ, DMODEL, DMODEL);

  dim3 g_in((2 * DINNER) / 256, L_SEQ / 64);
  k_gemm_f16_big<<<g_in, TB, 0, stream>>>(h_emb, wt_in, nullptr, xz, L_SEQ, 2 * DINNER, DMODEL);

  float* xs_f32 = (float*)fbufA;
  _Float16* xs_f16 = (_Float16*)hbufB;
  k_conv_silu<<<blocks(LD), TB, 0, stream>>>(xz, conv_w, conv_b, xs_f32, xs_f16);

  dim3 g_xp(96 / 32, L_SEQ / 128);
  k_gemm_f16<<<g_xp, TB, 0, stream>>>(xs_f16, wt_xp, nullptr, xdbl, L_SEQ, 96, DINNER);

  k_cvt_f16<<<blocks((size_t)L_SEQ * DTRANK), TB, 0, stream>>>(xdbl, h_dtin, L_SEQ, 96, DTRANK);

  float* dt = (float*)fbufB;
  dim3 g_dt(DINNER / 256, L_SEQ / 64);
  k_gemm_f16_big<<<g_dt, TB, 0, stream>>>(h_dtin, wt_dt, b_dt, dt, L_SEQ, DINNER, DTRANK);
  k_softplus<<<blocks(LD), TB, 0, stream>>>(dt, LD);

  float* ys = (float*)fbufC;
  k_scan<<<(DINNER * DSTATE) / TB, TB, 0, stream>>>(xs_f32, dt, xdbl, A_log, Dvec, ys);

  _Float16* h_gated = (_Float16*)hbufB;   // xs_f16 dead
  k_gate_silu<<<blocks(LD), TB, 0, stream>>>(ys, xz, h_gated);

  float* mamba_out = (float*)fbufB;                    // dt dead
  float* seq1      = (float*)(fbufB + LM * 4);
  dim3 g_op(DMODEL / 256, L_SEQ / 64);
  k_gemm_f16_big<<<g_op, TB, 0, stream>>>(h_gated, wt_out, nullptr, mamba_out, L_SEQ, DMODEL, DINNER);

  k_add_ln<<<L_SEQ, TB, 0, stream>>>(mamba_out, emb, ln1_g, ln1_b, seq1, nullptr, DMODEL, 0);

  // ---- cross-attention: softmax over size-1 axis == 1.0 => broadcast of
  //      attn_out(Wv @ clinical + bv); q/k are dead code; weights scalar = 1.0
  k_gemv<<<DMODEL / TB, TB, 0, stream>>>(w_ain + (size_t)2 * DMODEL * DMODEL, clin,
                                         b_ain + 2 * DMODEL, vvec, DMODEL, DMODEL);
  k_gemv<<<DMODEL / TB, TB, 0, stream>>>(w_aout, vvec, b_aout, avec, DMODEL, DMODEL);

  _Float16* h_seq2 = (_Float16*)hbufA;    // h_emb dead
  k_add_ln<<<L_SEQ, TB, 0, stream>>>(avec, seq1, ln2_g, ln2_b, seq2, h_seq2, DMODEL, 1);

  // ---- FFN ----
  float* ffn_lin1 = (float*)fbufA;        // xs_f32 dead
  dim3 g_f1((2 * DMODEL) / 256, L_SEQ / 64);
  k_gemm_f16_big<<<g_f1, TB, 0, stream>>>(h_seq2, wt_f1, b_f1, ffn_lin1, L_SEQ, 2 * DMODEL, DMODEL);

  _Float16* h_ffn = (_Float16*)fbufC;     // ys dead
  k_gelu<<<blocks(LD), TB, 0, stream>>>(ffn_lin1, h_ffn, LD);

  float* ffn_out = (float*)(fbufC + LM * 4);
  dim3 g_f2(DMODEL / 256, L_SEQ / 64);
  k_gemm_f16_big<<<g_f2, TB, 0, stream>>>(h_ffn, wt_f2, b_f2, ffn_out, L_SEQ, DMODEL, 2 * DMODEL);

  k_add_ln<<<L_SEQ, TB, 0, stream>>>(ffn_out, seq2, ln3_g, ln3_b, out, nullptr, DMODEL, 0);

  // weights output = mean of softmax over singleton axis = exactly 1.0
  k_store_one<<<1, 1, 0, stream>>>(out + (size_t)L_SEQ * DMODEL, 1.0f);

  (void)in_sizes; (void)n_in; (void)out_size; (void)ws_size;
}
